// Attention_773094113484
// MI455X (gfx1250) — matrix operand order repaired and checked by reading, not compiled, not run on hardware
//
#include <hip/hip_runtime.h>
#include <stdint.h>

// Problem constants
#define BATCH 64
#define LC    1024
#define LQ    128
#define HD    512
#define OH    (6 * HD)
#define NEGF  (-1e30f)

typedef __attribute__((ext_vector_type(16))) __bf16 v16bf_t;
typedef __attribute__((ext_vector_type(8)))  float  v8f_t;

__device__ __forceinline__ __bf16 tobf(float x) { return (__bf16)x; }

// ---------------------------------------------------------------------------
// Async global->LDS copy (CDNA5 GLOBAL_LOAD_ASYNC_TO_LDS_*, ASYNCcnt-tracked)
// Builtin signature (from clang diagnostics): param0 = AS1 (global) int/int4*,
// param1 = AS3 (LDS) int/int4*, then two immediate ints (offset, cpol).
// ---------------------------------------------------------------------------
typedef __attribute__((ext_vector_type(4))) int i4_t;
typedef __attribute__((address_space(1))) i4_t gi4_t;
typedef __attribute__((address_space(3))) i4_t li4_t;
typedef __attribute__((address_space(1))) int  gi1_t;
typedef __attribute__((address_space(3))) int  li1_t;

__device__ __forceinline__ void async_ld_b128(const float* g, float* l) {
#if __has_builtin(__builtin_amdgcn_global_load_async_to_lds_b128)
  __builtin_amdgcn_global_load_async_to_lds_b128(
      (gi4_t*)(uintptr_t)g, (li4_t*)(uint32_t)(uintptr_t)l, 0, 0);
#else
  asm volatile("global_load_async_to_lds_b128 %0, %1, off"
               :: "v"((uint32_t)(uintptr_t)l), "v"((uint64_t)(uintptr_t)g)
               : "memory");
#endif
}

__device__ __forceinline__ void async_ld_b32(const float* g, float* l) {
#if __has_builtin(__builtin_amdgcn_global_load_async_to_lds_b32)
  __builtin_amdgcn_global_load_async_to_lds_b32(
      (gi1_t*)(uintptr_t)g, (li1_t*)(uint32_t)(uintptr_t)l, 0, 0);
#else
  asm volatile("global_load_async_to_lds_b32 %0, %1, off"
               :: "v"((uint32_t)(uintptr_t)l), "v"((uint64_t)(uintptr_t)g)
               : "memory");
#endif
}

#if __has_builtin(__builtin_amdgcn_s_wait_asynccnt)
#define WAIT_ASYNC(n) __builtin_amdgcn_s_wait_asynccnt(n)
#else
#define WAIT_ASYNC(n) asm volatile("s_wait_asynccnt %0" :: "i"(n) : "memory")
#endif

// ---------------------------------------------------------------------------
// bf16-WMMA GEMM: D = epi(A[MxK] @ B[KxN])
//   BT  : 0 -> B row-major [K,N] (async b32 transpose-scatter into LDS)
//         1 -> B stored [N,K] (A@B^T form; async b128 direct copy)
//   EPI : 0 plain ; 1 sim (+row+col+bias) ; 2 bias+relu ;
//         3 triple-store (D2=c, D=v, D3=c*v) ; 4 D=c*v
//   ASC : 1 -> scale A element by ascale[k] during load (sync staging)
// Block: 128 threads = 4 wave32; block tile 64x64; wave tile 32x32 (2x2 WMMA);
// K-step 32; double-buffered LDS fed by ASYNCcnt-pipelined copies.
// ---------------------------------------------------------------------------
#define BM 64
#define BN 64
#define TK 32
#define LDSP 36  // padded LDS row stride (floats): 16B-aligned, bank-skewed

template <int BT, int EPI, int ASC>
__global__ __launch_bounds__(128) void gemm_bf16(
    const float* __restrict__ A, long sAb, int lda,
    const float* __restrict__ Bm, long sBb, int ldb,
    float* __restrict__ D, long sDb, int ldd,
    int M, int N, int K,
    const float* __restrict__ ascale,
    const float* __restrict__ radd, int sradd,
    const float* __restrict__ cadd, int scadd,
    const float* __restrict__ bias,
    const float* __restrict__ Cm, long sCb, int ldc,
    float* __restrict__ D2, long sD2b,
    float* __restrict__ D3, long sD3b)
{
  __shared__ float sA[2][BM * LDSP];  // sA[buf][m][k]
  __shared__ float sB[2][BN * LDSP];  // sB[buf][n][k]

  const int b  = blockIdx.z;
  const int m0 = blockIdx.x * BM;
  const int n0 = blockIdx.y * BN;
  A  += (long)b * sAb;
  Bm += (long)b * sBb;
  D  += (long)b * sDb;

  const int tid  = threadIdx.x;
  const int lane = tid & 31;
  const int wave = tid >> 5;
  const int wm   = (wave & 1) * 32;   // wave tile origin in block
  const int wn   = (wave >> 1) * 32;
  const int h    = lane >> 4;         // half-wave
  const int l15  = lane & 15;

  const int lr  = tid >> 3;           // 0..15: row per load pass
  const int lc4 = (tid & 7) * 4;      // 4-float column group

  // async ops issued per thread per K-tile (immediate for s_wait_asynccnt)
  constexpr int LTOT = (ASC ? 0 : 4) + (BT ? 4 : 16);

  v8f_t acc[2][2] = {};

  auto issue = [&](int k0, int buf) {
    // ---- A tile (BM x TK) ----
    if (ASC) {
#pragma unroll
      for (int p = 0; p < 4; ++p) {
        int r = lr + 16 * p;
        float4 v = *(const float4*)(A + (long)(m0 + r) * lda + (k0 + lc4));
        v.x *= ascale[k0 + lc4 + 0];
        v.y *= ascale[k0 + lc4 + 1];
        v.z *= ascale[k0 + lc4 + 2];
        v.w *= ascale[k0 + lc4 + 3];
        *(float4*)&sA[buf][r * LDSP + lc4] = v;
      }
    } else {
#pragma unroll
      for (int p = 0; p < 4; ++p) {
        int r = lr + 16 * p;
        async_ld_b128(A + (long)(m0 + r) * lda + (k0 + lc4),
                      &sA[buf][r * LDSP + lc4]);
      }
    }
    // ---- B tile -> LDS [n][k] ----
    if (BT == 1) {
      // Bmem [N,K]: contiguous along k, direct async b128 copy
#pragma unroll
      for (int p = 0; p < 4; ++p) {
        int n = lr + 16 * p;
        async_ld_b128(Bm + (long)(n0 + n) * ldb + (k0 + lc4),
                      &sB[buf][n * LDSP + lc4]);
      }
    } else {
      // Bmem [K,N]: coalesced along n, per-lane async b32 transpose-scatter
#pragma unroll
      for (int j = 0; j < 16; ++j) {
        int id = tid + 128 * j;        // 2048 = 32k x 64n elements
        int kk = id >> 6;
        int nn = id & 63;
        async_ld_b32(Bm + (long)(k0 + kk) * ldb + (n0 + nn),
                     &sB[buf][nn * LDSP + kk]);
      }
    }
  };

  const int nk = K / TK;
  issue(0, 0);
  for (int i = 0; i < nk; ++i) {
    if (i + 1 < nk) {
      issue((i + 1) * TK, (i + 1) & 1);
      WAIT_ASYNC(LTOT);   // tile i complete (per-wave in-order), i+1 in flight
    } else {
      WAIT_ASYNC(0);
    }
    __syncthreads();

    const int buf = i & 1;
    const float* a0 = &sA[buf][(wm + l15) * LDSP];
    const float* a1 = a0 + 16 * LDSP;
    const float* b0 = &sB[buf][(wn + l15) * LDSP];
    const float* b1 = b0 + 16 * LDSP;

    // fragments per ISA 7.12.2 16-bit layouts:
    // A: K = e+8h (e<8), e+8+8h (e>=8) ; B: same
    v16bf_t af0, af1, bf0, bf1;
#pragma unroll
    for (int e = 0; e < 8; ++e) {
      af0[e]     = tobf(a0[8 * h + e]);
      af0[8 + e] = tobf(a0[16 + 8 * h + e]);
      af1[e]     = tobf(a1[8 * h + e]);
      af1[8 + e] = tobf(a1[16 + 8 * h + e]);
    }
#pragma unroll
    for (int e = 0; e < 16; ++e) {
      bf0[e] = tobf(b0[(e & 8) * 2 + 8 * h + (e & 7)]);
      bf1[e] = tobf(b1[(e & 8) * 2 + 8 * h + (e & 7)]);
    }

    acc[0][0] = __builtin_amdgcn_wmma_f32_16x16x32_bf16(
        false, af0, false, bf0, (short)0, acc[0][0], false, false);
    acc[0][1] = __builtin_amdgcn_wmma_f32_16x16x32_bf16(
        false, af0, false, bf1, (short)0, acc[0][1], false, false);
    acc[1][0] = __builtin_amdgcn_wmma_f32_16x16x32_bf16(
        false, af1, false, bf0, (short)0, acc[1][0], false, false);
    acc[1][1] = __builtin_amdgcn_wmma_f32_16x16x32_bf16(
        false, af1, false, bf1, (short)0, acc[1][1], false, false);

    __syncthreads();
  }

  // ---- epilogue: C/D element r -> M = 8h + r, N = lane&15 ----
#pragma unroll
  for (int tm = 0; tm < 2; ++tm) {
#pragma unroll
    for (int tn = 0; tn < 2; ++tn) {
      const int n = n0 + wn + 16 * tn + l15;
#pragma unroll
      for (int r = 0; r < 8; ++r) {
        const int m = m0 + wm + 16 * tm + 8 * h + r;
        float v = acc[tm][tn][r];
        const long idx = (long)m * ldd + n;
        if (EPI == 1) {
          v += radd[(long)b * sradd + m] + cadd[(long)b * scadd + n] + bias[0];
          D[idx] = v;
        } else if (EPI == 2) {
          v += bias[n];
          D[idx] = v > 0.f ? v : 0.f;
        } else if (EPI == 3) {
          float cv = Cm[(long)b * sCb + (long)m * ldc + n];
          D2[(long)b * sD2b + idx] = cv;      // segment: c
          D[idx] = v;                         // segment: a
          D3[(long)b * sD3b + idx] = cv * v;  // segment: c*a
        } else if (EPI == 4) {
          float cv = Cm[(long)b * sCb + (long)m * ldc + n];
          D[idx] = cv * v;                    // segment: c*b
        } else {
          D[idx] = v;
        }
      }
    }
  }
}

// ---------------------------------------------------------------------------
// s0/s1: per-row dot with a weight vector. One wave per row.
// ---------------------------------------------------------------------------
__global__ __launch_bounds__(128) void rowdot_kernel(
    const float* __restrict__ X, const float* __restrict__ w,
    float* __restrict__ out, int rows, int Kdim)
{
  int row  = blockIdx.x * 4 + (threadIdx.x >> 5);
  int lane = threadIdx.x & 31;
  if (row >= rows) return;
  const float* x = X + (long)row * Kdim;
  float s = 0.f;
  for (int k = lane; k < Kdim; k += 32) s += x[k] * w[k];
#pragma unroll
  for (int off = 16; off; off >>= 1) s += __shfl_xor(s, off, 32);
  if (lane == 0) out[row] = s;
}

// ---------------------------------------------------------------------------
// Row softmax over LQ=128 with q_mask. One wave per row, 4 elems/lane.
// ---------------------------------------------------------------------------
__global__ __launch_bounds__(128) void row_softmax_kernel(
    const float* __restrict__ S, const int* __restrict__ qmask,
    float* __restrict__ P, int rows_per_batch)
{
  int row  = blockIdx.x * 4 + (threadIdx.x >> 5);
  int lane = threadIdx.x & 31;
  int b    = row / rows_per_batch;
  const float* s  = S + (long)row * LQ;
  const int*   qm = qmask + (long)b * LQ;
  float v[4];
#pragma unroll
  for (int j = 0; j < 4; ++j) {
    int n = lane + 32 * j;
    v[j] = (qm[n] != 0) ? s[n] : NEGF;
  }
  float mx = fmaxf(fmaxf(v[0], v[1]), fmaxf(v[2], v[3]));
#pragma unroll
  for (int off = 16; off; off >>= 1) mx = fmaxf(mx, __shfl_xor(mx, off, 32));
  float sum = 0.f;
#pragma unroll
  for (int j = 0; j < 4; ++j) { v[j] = __expf(v[j] - mx); sum += v[j]; }
#pragma unroll
  for (int off = 16; off; off >>= 1) sum += __shfl_xor(sum, off, 32);
  float inv = 1.f / sum;
  float* p = P + (long)row * LQ;
#pragma unroll
  for (int j = 0; j < 4; ++j) p[lane + 32 * j] = v[j] * inv;
}

// ---------------------------------------------------------------------------
// Column softmax over LC=1024 with c_mask; output TRANSPOSED: PT[q][c].
// One 256-thread block per (q, b); 4 elems/thread.
// ---------------------------------------------------------------------------
__global__ __launch_bounds__(256) void col_softmax_kernel(
    const float* __restrict__ S, const int* __restrict__ cmask,
    float* __restrict__ PT)
{
  const int q = blockIdx.x;
  const int b = blockIdx.y;
  const float* s  = S + (long)b * LC * LQ + q;
  const int*   cm = cmask + (long)b * LC;
  float* pt = PT + ((long)b * LQ + q) * LC;

  const int tid = threadIdx.x;
  float v[4];
#pragma unroll
  for (int i = 0; i < 4; ++i) {
    int c = tid + 256 * i;
    v[i] = (cm[c] != 0) ? s[(long)c * LQ] : NEGF;
  }
  __shared__ float redmax[8];
  __shared__ float redsum[8];
  float mx = fmaxf(fmaxf(v[0], v[1]), fmaxf(v[2], v[3]));
#pragma unroll
  for (int off = 16; off; off >>= 1) mx = fmaxf(mx, __shfl_xor(mx, off, 32));
  if ((tid & 31) == 0) redmax[tid >> 5] = mx;
  __syncthreads();
  float bm = redmax[0];
#pragma unroll
  for (int i = 1; i < 8; ++i) bm = fmaxf(bm, redmax[i]);
  float sum = 0.f;
#pragma unroll
  for (int i = 0; i < 4; ++i) { v[i] = __expf(v[i] - bm); sum += v[i]; }
#pragma unroll
  for (int off = 16; off; off >>= 1) sum += __shfl_xor(sum, off, 32);
  if ((tid & 31) == 0) redsum[tid >> 5] = sum;
  __syncthreads();
  float ts = 0.f;
#pragma unroll
  for (int i = 0; i < 8; ++i) ts += redsum[i];
  float inv = 1.f / ts;
#pragma unroll
  for (int i = 0; i < 4; ++i) pt[tid + 256 * i] = v[i] * inv;
}

// ---------------------------------------------------------------------------
extern "C" void kernel_launch(void* const* d_in, const int* in_sizes, int n_in,
                              void* d_out, int out_size, void* d_ws, size_t ws_size,
                              hipStream_t stream)
{
  const float* c    = (const float*)d_in[0];
  const float* q    = (const float*)d_in[1];
  const int*   cmsk = (const int*)d_in[2];
  const int*   qmsk = (const int*)d_in[3];
  const float* c_w  = (const float*)d_in[4];
  const float* q_w  = (const float*)d_in[5];
  const float* cq_w = (const float*)d_in[6];
  const float* bias = (const float*)d_in[7];
  const float* W1   = (const float*)d_in[8];
  const float* b1   = (const float*)d_in[9];
  const float* W2   = (const float*)d_in[10];
  const float* b2   = (const float*)d_in[11];
  float* out = (float*)d_out;

  // workspace layout (floats), ~235 MB total
  float* f    = (float*)d_ws;
  float* s0   = f;                         // [B, LC]
  float* s1   = s0  + (size_t)BATCH * LC;  // [B, LQ]
  float* sim  = s1  + (size_t)BATCH * LQ;  // [B, LC, LQ]  (reused for scoat)
  float* p1   = sim + (size_t)BATCH * LC * LQ;  // s1s
  float* p2T  = p1  + (size_t)BATCH * LC * LQ;  // s2s transposed [B, LQ, LC]
  float* tbuf = p2T + (size_t)BATCH * LC * LQ;  // s2s^T @ c : [B, LQ, H]
  float* h1   = tbuf + (size_t)BATCH * LQ * HD; // relu(q@W1+b1)
  float* qp   = h1  + (size_t)BATCH * LQ * HD;  // [B, LQ, H]
  float* sc1  = qp  + (size_t)BATCH * LQ * HD;  // [B, LC, LQ]
  float* sc2T = sc1 + (size_t)BATCH * LC * LQ;  // [B, LQ, LC]
  float* bco  = sc2T + (size_t)BATCH * LC * LQ; // bcoat [B, LQ, H]

  const long sC = (long)LC * HD, sQ = (long)LQ * HD;
  const long sS = (long)LC * LQ, sST = (long)LQ * LC;
  const long sO = (long)LC * OH, sT = (long)LQ * HD;

  // 1) s0 = c @ c_weight ; s1 = q @ q_weight
  rowdot_kernel<<<BATCH * LC / 4, 128, 0, stream>>>(c, c_w, s0, BATCH * LC, HD);
  rowdot_kernel<<<BATCH * LQ / 4, 128, 0, stream>>>(q, q_w, s1, BATCH * LQ, HD);

  // 2) sim = (c*cq_w) @ q^T + s0 + s1 + bias
  gemm_bf16<1, 1, 1><<<dim3(LC / BM, LQ / BN, BATCH), 128, 0, stream>>>(
      c, sC, HD, q, sQ, HD, sim, sS, LQ, LC, LQ, HD,
      cq_w, s0, LC, s1, LQ, bias, nullptr, 0, 0, nullptr, 0, nullptr, 0);

  // 3) dual masked softmax
  row_softmax_kernel<<<BATCH * LC / 4, 128, 0, stream>>>(sim, qmsk, p1, LC);
  col_softmax_kernel<<<dim3(LQ, BATCH), 256, 0, stream>>>(sim, cmsk, p2T);

  // 4) t = s2s^T @ c
  gemm_bf16<0, 0, 0><<<dim3(LQ / BM, HD / BN, BATCH), 128, 0, stream>>>(
      p2T, sST, LC, c, sC, HD, tbuf, sT, HD, LQ, HD, LC,
      nullptr, nullptr, 0, nullptr, 0, nullptr, nullptr, 0, 0, nullptr, 0, nullptr, 0);

  // 5) a = s1s @ q ; fused stores of c (seg0), a (seg1), c*a (seg2)
  gemm_bf16<0, 3, 0><<<dim3(LC / BM, HD / BN, BATCH), 128, 0, stream>>>(
      p1, sS, LQ, q, sQ, HD, out + HD, sO, OH, LC, HD, LQ,
      nullptr, nullptr, 0, nullptr, 0, nullptr, c, sC, HD,
      out, sO, out + 2 * HD, sO);

  // 6) c*b = c * (s1s @ t) -> seg3
  gemm_bf16<0, 4, 0><<<dim3(LC / BM, HD / BN, BATCH), 128, 0, stream>>>(
      p1, sS, LQ, tbuf, sT, HD, out + 3 * HD, sO, OH, LC, HD, LQ,
      nullptr, nullptr, 0, nullptr, 0, nullptr, c, sC, HD, nullptr, 0, nullptr, 0);

  // 7) MLP: h1 = relu(q@W1+b1) ; qp = relu(h1@W2+b2)
  gemm_bf16<0, 2, 0><<<dim3(LQ / BM, HD / BN, BATCH), 128, 0, stream>>>(
      q, sQ, HD, W1, 0, HD, h1, sT, HD, LQ, HD, HD,
      nullptr, nullptr, 0, nullptr, 0, b1, nullptr, 0, 0, nullptr, 0, nullptr, 0);
  gemm_bf16<0, 2, 0><<<dim3(LQ / BM, HD / BN, BATCH), 128, 0, stream>>>(
      h1, sT, HD, W2, 0, HD, qp, sT, HD, LQ, HD, HD,
      nullptr, nullptr, 0, nullptr, 0, b2, nullptr, 0, 0, nullptr, 0, nullptr, 0);

  // 8) scoat = c @ qp^T  (reuse sim buffer)
  gemm_bf16<1, 0, 0><<<dim3(LC / BM, LQ / BN, BATCH), 128, 0, stream>>>(
      c, sC, HD, qp, sT, HD, sim, sS, LQ, LC, LQ, HD,
      nullptr, nullptr, 0, nullptr, 0, nullptr, nullptr, 0, 0, nullptr, 0, nullptr, 0);

  // 9) coattention softmaxes
  row_softmax_kernel<<<BATCH * LC / 4, 128, 0, stream>>>(sim, qmsk, sc1, LC);
  col_softmax_kernel<<<dim3(LQ, BATCH), 256, 0, stream>>>(sim, cmsk, sc2T);

  // 10) bcoat = sc2^T @ c
  gemm_bf16<0, 0, 0><<<dim3(LQ / BM, HD / BN, BATCH), 128, 0, stream>>>(
      sc2T, sST, LC, c, sC, HD, bco, sT, HD, LQ, HD, LC,
      nullptr, nullptr, 0, nullptr, 0, nullptr, nullptr, 0, 0, nullptr, 0, nullptr, 0);

  // 11) scoat3 = sc1 @ bcoat -> seg4
  gemm_bf16<0, 0, 0><<<dim3(LC / BM, HD / BN, BATCH), 128, 0, stream>>>(
      sc1, sS, LQ, bco, sT, HD, out + 4 * HD, sO, OH, LC, HD, LQ,
      nullptr, nullptr, 0, nullptr, 0, nullptr, nullptr, 0, 0, nullptr, 0, nullptr, 0);

  // 12) acoat = sc1 @ qp -> seg5
  gemm_bf16<0, 0, 0><<<dim3(LC / BM, HD / BN, BATCH), 128, 0, stream>>>(
      sc1, sS, LQ, qp, sT, HD, out + 5 * HD, sO, OH, LC, HD, LQ,
      nullptr, nullptr, 0, nullptr, 0, nullptr, nullptr, 0, 0, nullptr, 0, nullptr, 0);

  (void)in_sizes; (void)n_in; (void)out_size; (void)ws_size;
}
